// MyTransformer_74345883894159
// MI455X (gfx1250) — matrix operand order repaired and checked by reading, not compiled, not run on hardware
//
#include <hip/hip_runtime.h>
#include <hip/hip_bf16.h>

// ---------------------------------------------------------------------------
// MI455X (gfx1250) transformer block: embed + GQA attention + out-proj.
// All matmuls via v_wmma_f32_16x16x32_f16 (f16 in, f32 accum), wave32.
// GEMMs: 128-thread blocks, 128x64 block tile, 32x64 per-wave tile,
// double-buffered LDS (one barrier per k-step, 8 WMMA per 12 ds loads).
// ---------------------------------------------------------------------------

typedef _Float16 f16;
typedef _Float16 half4_t __attribute__((ext_vector_type(4)));
typedef _Float16 half8_t __attribute__((ext_vector_type(8)));
typedef _Float16 v16h    __attribute__((ext_vector_type(16)));
typedef float    v8f     __attribute__((ext_vector_type(8)));

#define D_MODEL 1024
#define T_CTX   2048
#define NB      4
#define NHEAD   16
#define NKV     4
#define HD      64
#define MTOT    (NB * T_CTX)   /* 8192 rows of x */
#define NQKV    1536           /* 1024 Q + 256 K + 256 V output cols */
#define LDSW    40             /* padded LDS row stride in halves (80B) */

static __device__ __forceinline__ v16h mk16(half8_t lo, half8_t hi) {
  return __builtin_shufflevector(lo, hi, 0,1,2,3,4,5,6,7,8,9,10,11,12,13,14,15);
}
static __device__ __forceinline__ v8f wmma_f16(v16h a, v16h b, v8f c) {
  // (neg_a, A, neg_b, B, c_mod, C, reuse_a, reuse_b)
  return __builtin_amdgcn_wmma_f32_16x16x32_f16(false, a, false, b, (short)0, c,
                                                false, false);
}

// ---------------------------------------------------------------------------
// 1) Embedding: xh[m, d] = d<512 ? tok[idx[m]][d] : pos[t][d-512]   (f32->f16)
// ---------------------------------------------------------------------------
__global__ __launch_bounds__(256) void embed_kernel(
    const int* __restrict__ idx, const float* __restrict__ tok,
    const float* __restrict__ pos, f16* __restrict__ xh) {
  int m = blockIdx.x;              // 0..8191
  int d = threadIdx.x * 4;         // 0..1020
  int t = m & (T_CTX - 1);
  const float* src = (d < 512) ? (tok + (size_t)idx[m] * 512 + d)
                               : (pos + (size_t)t * 512 + (d - 512));
  float4 v = *(const float4*)src;
  half4_t h;
  h[0] = (f16)v.x; h[1] = (f16)v.y; h[2] = (f16)v.z; h[3] = (f16)v.w;
  *(half4_t*)(xh + (size_t)m * D_MODEL + d) = h;
}

// ---------------------------------------------------------------------------
// 2) Pack weights: Wqkvt[n][k] = [Wq|Wk|Wv][k][n] (f16),  Wot[n][k] = Wo[k][n]
//    Transposed so a WMMA B-fragment (one output column) is contiguous in K.
// ---------------------------------------------------------------------------
__global__ __launch_bounds__(256) void pack_weights_kernel(
    const float* __restrict__ Wq, const float* __restrict__ Wk,
    const float* __restrict__ Wv, const float* __restrict__ Wo,
    f16* __restrict__ Wqkvt, f16* __restrict__ Wot) {
  int id = blockIdx.x * 256 + threadIdx.x;
  const int tot1 = NQKV * D_MODEL;
  if (id < tot1) {
    int n = id >> 10, k = id & 1023;
    float v;
    if (n < 1024)      v = Wq[(size_t)k * 1024 + n];
    else if (n < 1280) v = Wk[(size_t)k * 256 + (n - 1024)];
    else               v = Wv[(size_t)k * 256 + (n - 1280)];
    Wqkvt[id] = (f16)v;
  } else if (id < tot1 + D_MODEL * D_MODEL) {
    int j = id - tot1;
    int n = j >> 10, k = j & 1023;
    Wot[j] = (f16)Wo[(size_t)k * 1024 + n];
  }
}

// ---------------------------------------------------------------------------
// Shared GEMM core: 4 waves, block tile 128(M) x 64(N), K-step 32,
// double-buffered LDS. Each wave: 2 row-tiles x 4 col-tiles accumulators.
// Epilogue differs (QKV scatter vs f32 out), so two kernels share this macro.
// ---------------------------------------------------------------------------
#define GEMM_CORE(Aptr, Btptr)                                                  \
  __shared__ f16 As[2][128 * LDSW];                                             \
  __shared__ f16 Bs[2][64 * LDSW];                                              \
  const int m0 = blockIdx.x * 128;                                              \
  const int n0 = blockIdx.y * 64;                                               \
  const int tid = threadIdx.x;                                                  \
  const int wave = tid >> 5, ln = tid & 31;                                     \
  const int hs = ln >> 4, lr = ln & 15;                                         \
  const int ko = hs * 8;   /* A-frag K base (ISA 16-bit A layout) */            \
  const int kb = hs * 8;   /* B-frag K base (ISA 16-bit B layout) */            \
  const int cr = tid >> 2, cseg = (tid & 3) * 8; /* copy row/segment */         \
  v8f acc[2][4] = {};                                                           \
  /* prologue: fill buffer 0 with K-slab 0 */                                   \
  _Pragma("unroll") for (int it = 0; it < 4; ++it) {                            \
    int r = cr + it * 32;                                                       \
    *(half8_t*)&As[0][r * LDSW + cseg] =                                        \
        *(const half8_t*)&Aptr[(size_t)(m0 + r) * D_MODEL + cseg];              \
  }                                                                             \
  _Pragma("unroll") for (int it = 0; it < 2; ++it) {                            \
    int r = cr + it * 32;                                                       \
    *(half8_t*)&Bs[0][r * LDSW + cseg] =                                        \
        *(const half8_t*)&Btptr[(size_t)(n0 + r) * D_MODEL + cseg];             \
  }                                                                             \
  __syncthreads();                                                              \
  int buf = 0;                                                                  \
  for (int kk = 0; kk < D_MODEL; kk += 32, buf ^= 1) {                          \
    if (kk + 64 < D_MODEL) { /* L2 prefetch two slabs ahead */                  \
      __builtin_prefetch(&Aptr[(size_t)(m0 + cr) * D_MODEL + kk + 64 + cseg], 0, 3); \
      __builtin_prefetch(&Btptr[(size_t)(n0 + (cr & 63)) * D_MODEL + kk + 64 + cseg], 0, 3); \
    }                                                                           \
    if (kk + 32 < D_MODEL) { /* stage next slab into the other buffer */        \
      _Pragma("unroll") for (int it = 0; it < 4; ++it) {                        \
        int r = cr + it * 32;                                                   \
        *(half8_t*)&As[buf ^ 1][r * LDSW + cseg] =                              \
            *(const half8_t*)&Aptr[(size_t)(m0 + r) * D_MODEL + kk + 32 + cseg]; \
      }                                                                         \
      _Pragma("unroll") for (int it = 0; it < 2; ++it) {                        \
        int r = cr + it * 32;                                                   \
        *(half8_t*)&Bs[buf ^ 1][r * LDSW + cseg] =                              \
            *(const half8_t*)&Btptr[(size_t)(n0 + r) * D_MODEL + kk + 32 + cseg]; \
      }                                                                         \
    }                                                                           \
    /* fragments from current buffer */                                         \
    int ra0 = wave * 32 + lr, ra1 = ra0 + 16;                                   \
    v16h a0 = mk16(*(const half8_t*)&As[buf][ra0 * LDSW + ko],                  \
                   *(const half8_t*)&As[buf][ra0 * LDSW + ko + 16]);            \
    v16h a1 = mk16(*(const half8_t*)&As[buf][ra1 * LDSW + ko],                  \
                   *(const half8_t*)&As[buf][ra1 * LDSW + ko + 16]);            \
    _Pragma("unroll") for (int n = 0; n < 4; ++n) {                             \
      int rb = n * 16 + lr;                                                     \
      v16h bf = mk16(*(const half8_t*)&Bs[buf][rb * LDSW + kb],                 \
                     *(const half8_t*)&Bs[buf][rb * LDSW + kb + 16]);           \
      acc[0][n] = wmma_f16(a0, bf, acc[0][n]);                                  \
      acc[1][n] = wmma_f16(a1, bf, acc[1][n]);                                  \
    }                                                                           \
    __syncthreads();                                                            \
  }

// ---------------------------------------------------------------------------
// 3) QKV GEMM: [8192,1024] x [1024,1536].
//    Scatters: Qh[b][h][t][d], Kh[b][g][t][d], Vt[b][g][d][t] (transposed).
// ---------------------------------------------------------------------------
__global__ __launch_bounds__(128) void gemm_qkv_kernel(
    const f16* __restrict__ A, const f16* __restrict__ Bt,
    const float* __restrict__ bq, const float* __restrict__ bk,
    const float* __restrict__ bv,
    f16* __restrict__ Qh, f16* __restrict__ Kh, f16* __restrict__ Vt) {
  GEMM_CORE(A, Bt)
#pragma unroll
  for (int n = 0; n < 4; ++n) {
    int ncol = n0 + n * 16 + lr;
    float bias = (ncol < 1024) ? bq[ncol]
               : (ncol < 1280) ? bk[ncol - 1024] : bv[ncol - 1280];
#pragma unroll
    for (int rt = 0; rt < 2; ++rt)
#pragma unroll
      for (int i = 0; i < 8; ++i) {
        int m = m0 + wave * 32 + rt * 16 + i + hs * 8;  // C-tile row mapping
        int b = m >> 11, t = m & (T_CTX - 1);
        f16 hv = (f16)(acc[rt][n][i] + bias);
        if (ncol < 1024) {
          int hh = ncol >> 6, dn = ncol & 63;
          Qh[(((size_t)b * NHEAD + hh) * T_CTX + t) * HD + dn] = hv;
        } else if (ncol < 1280) {
          int g = (ncol - 1024) >> 6, dn = ncol & 63;
          Kh[(((size_t)b * NKV + g) * T_CTX + t) * HD + dn] = hv;
        } else {
          int g = (ncol - 1280) >> 6, dn = ncol & 63;
          Vt[(((size_t)b * NKV + g) * HD + dn) * T_CTX + t] = hv;  // transposed
        }
      }
  }
}

// ---------------------------------------------------------------------------
// 4) Flash attention, one wave per (b, h, 16-row q tile), causal, GQA g=h/4.
//    Per 32-key chunk: 4 WMMA for Q.K^T, online softmax (shfl_xor over the
//    16-lane halves), P -> f16 via LDS re-layout, 4 WMMA for P.V.
//    K/V fragments load straight from global: K+V = 8 MB, L2-resident.
// ---------------------------------------------------------------------------
__global__ __launch_bounds__(32) void attn_kernel(
    const f16* __restrict__ Qh, const f16* __restrict__ Kh,
    const f16* __restrict__ Vt, f16* __restrict__ attnh) {
  const int qt = blockIdx.x, h = blockIdx.y, b = blockIdx.z;
  const int g = h >> 2;
  const int ln = threadIdx.x;
  const int hs = ln >> 4, lr = ln & 15;
  const int koA = hs * 8, kbB = hs * 8;
  const int qs = qt * 16;

  const f16* Qp = Qh + ((size_t)b * NHEAD + h) * T_CTX * HD;
  const f16* Kp = Kh + ((size_t)b * NKV + g) * T_CTX * HD;
  const f16* Vp = Vt + ((size_t)b * NKV + g) * HD * T_CTX;

  // Q A-fragments for the two 32-wide d-chunks (loaded once)
  v16h qa[2];
#pragma unroll
  for (int c = 0; c < 2; ++c) {
    const f16* qrow = Qp + (size_t)(qs + lr) * HD + c * 32;
    qa[c] = mk16(*(const half8_t*)&qrow[koA], *(const half8_t*)&qrow[koA + 16]);
  }

  v8f O[4] = {};
  float mrow[8], lrow[8];
#pragma unroll
  for (int i = 0; i < 8; ++i) { mrow[i] = -1e30f; lrow[i] = 0.f; }

  __shared__ f16 Pt[16 * LDSW];

  const int kend = qs + 16;                 // causal: keys <= max row in tile
  for (int k0 = 0; k0 < kend; k0 += 32) {
    // --- S = Q . K^T over this 32-key chunk (two 16-key C tiles) ---
    v8f S[2] = {};
#pragma unroll
    for (int j = 0; j < 2; ++j) {
      int key = k0 + j * 16 + lr;
      const f16* krow = Kp + (size_t)key * HD;
#pragma unroll
      for (int c = 0; c < 2; ++c) {
        int kb = c * 32 + kbB;
        v16h kf = mk16(*(const half8_t*)&krow[kb], *(const half8_t*)&krow[kb + 16]);
        S[j] = wmma_f16(qa[c], kf, S[j]);
      }
    }
    // --- online softmax (rows live at VGPR index i, cols across 16 lanes) ---
    const float scale = 0.125f;             // 1/sqrt(64)
    float p0[8], p1[8], alpha[8];
#pragma unroll
    for (int i = 0; i < 8; ++i) {
      int row = qs + i + hs * 8;
      float s0 = S[0][i] * scale, s1 = S[1][i] * scale;
      if (k0 + lr > row)      s0 = -1e30f;  // causal mask
      if (k0 + 16 + lr > row) s1 = -1e30f;
      float t = fmaxf(s0, s1);
#pragma unroll
      for (int off = 8; off >= 1; off >>= 1)
        t = fmaxf(t, __shfl_xor(t, off, 32));
      float mnew = fmaxf(mrow[i], t);
      float al = __expf(mrow[i] - mnew);
      float e0 = __expf(s0 - mnew), e1 = __expf(s1 - mnew);
      float rs = e0 + e1;
#pragma unroll
      for (int off = 8; off >= 1; off >>= 1)
        rs += __shfl_xor(rs, off, 32);
      lrow[i] = lrow[i] * al + rs;
      mrow[i] = mnew;
      alpha[i] = al;
      p0[i] = e0; p1[i] = e1;
    }
#pragma unroll
    for (int n = 0; n < 4; ++n)
#pragma unroll
      for (int i = 0; i < 8; ++i) O[n][i] = O[n][i] * alpha[i];

    // --- C-layout P -> A-layout f16 fragment via LDS ---
    __syncthreads();
#pragma unroll
    for (int i = 0; i < 8; ++i) {
      int r = i + hs * 8;
      Pt[r * LDSW + lr]      = (f16)p0[i];
      Pt[r * LDSW + 16 + lr] = (f16)p1[i];
    }
    __syncthreads();
    v16h pa = mk16(*(const half8_t*)&Pt[lr * LDSW + koA],
                   *(const half8_t*)&Pt[lr * LDSW + koA + 16]);
    // --- O += P . V  (V stored transposed: column of B is contiguous) ---
#pragma unroll
    for (int n = 0; n < 4; ++n) {
      const f16* vrow = Vp + (size_t)(n * 16 + lr) * T_CTX + k0;
      v16h vf = mk16(*(const half8_t*)&vrow[kbB], *(const half8_t*)&vrow[kbB + 16]);
      O[n] = wmma_f16(pa, vf, O[n]);
    }
    __syncthreads();
  }

  // epilogue: normalize, write attnh[m][h*64+dn] (heads concatenated)
#pragma unroll
  for (int n = 0; n < 4; ++n)
#pragma unroll
    for (int i = 0; i < 8; ++i) {
      int row = qs + i + hs * 8;
      size_t m = (size_t)b * T_CTX + row;
      attnh[m * D_MODEL + h * HD + n * 16 + lr] = (f16)(O[n][i] / lrow[i]);
    }
}

// ---------------------------------------------------------------------------
// 5) Output projection: [8192,1024] x [1024,1024] + b_o -> f32 d_out
// ---------------------------------------------------------------------------
__global__ __launch_bounds__(128) void gemm_out_kernel(
    const f16* __restrict__ A, const f16* __restrict__ Bt,
    const float* __restrict__ bo, float* __restrict__ out) {
  GEMM_CORE(A, Bt)
#pragma unroll
  for (int n = 0; n < 4; ++n) {
    int ncol = n0 + n * 16 + lr;
    float bias = bo[ncol];
#pragma unroll
    for (int rt = 0; rt < 2; ++rt)
#pragma unroll
      for (int i = 0; i < 8; ++i) {
        int m = m0 + wave * 32 + rt * 16 + i + hs * 8;
        out[(size_t)m * D_MODEL + ncol] = acc[rt][n][i] + bias;
      }
  }
}

// ---------------------------------------------------------------------------
extern "C" void kernel_launch(void* const* d_in, const int* in_sizes, int n_in,
                              void* d_out, int out_size, void* d_ws, size_t ws_size,
                              hipStream_t stream) {
  (void)in_sizes; (void)n_in; (void)out_size; (void)ws_size;
  const int*   idx = (const int*)  d_in[0];
  // d_in[1] = mask: unused (causal mask applied analytically)
  const float* tok = (const float*)d_in[2];
  const float* pos = (const float*)d_in[3];
  const float* Wq  = (const float*)d_in[4];
  const float* bq  = (const float*)d_in[5];
  const float* Wk  = (const float*)d_in[6];
  const float* bk  = (const float*)d_in[7];
  const float* Wv  = (const float*)d_in[8];
  const float* bv  = (const float*)d_in[9];
  const float* Wo  = (const float*)d_in[10];
  const float* bo  = (const float*)d_in[11];
  float* out = (float*)d_out;

  // workspace layout (f16):  ~64 MB total
  f16* xh    = (f16*)d_ws;                          // 8192*1024
  f16* Wqkvt = xh    + (size_t)MTOT * D_MODEL;      // 1536*1024
  f16* Wot   = Wqkvt + (size_t)NQKV * D_MODEL;      // 1024*1024
  f16* Qh    = Wot   + (size_t)D_MODEL * D_MODEL;   // 8192*1024
  f16* Kh    = Qh    + (size_t)MTOT * D_MODEL;      // 2048*1024
  f16* Vt    = Kh    + (size_t)NB * NKV * T_CTX * HD;
  f16* attnh = Vt    + (size_t)NB * NKV * T_CTX * HD; // 8192*1024

  embed_kernel<<<MTOT, 256, 0, stream>>>(idx, tok, pos, xh);

  const int packTot = NQKV * D_MODEL + D_MODEL * D_MODEL;
  pack_weights_kernel<<<(packTot + 255) / 256, 256, 0, stream>>>(
      Wq, Wk, Wv, Wo, Wqkvt, Wot);

  gemm_qkv_kernel<<<dim3(MTOT / 128, NQKV / 64), 128, 0, stream>>>(
      xh, Wqkvt, bq, bk, bv, Qh, Kh, Vt);

  attn_kernel<<<dim3(T_CTX / 16, NHEAD, NB), 32, 0, stream>>>(Qh, Kh, Vt, attnh);

  gemm_out_kernel<<<dim3(MTOT / 128, D_MODEL / 64), 128, 0, stream>>>(
      attnh, Wot, bo, out);
}
